// RotaryMultiHeadAttention_58291296141597
// MI455X (gfx1250) — matrix operand order repaired and checked by reading, not compiled, not run on hardware
//
#include <hip/hip_runtime.h>
#include <hip/hip_bf16.h>
#include <math.h>

typedef __bf16 bf16_t;
typedef __attribute__((ext_vector_type(16))) __bf16 v16bf;
typedef __attribute__((ext_vector_type(8)))  __bf16 v8bf;
typedef __attribute__((ext_vector_type(8)))  float  v8f;
typedef int v4i __attribute__((vector_size(4 * sizeof(int))));   // matches builtin param

#define EMBED 1024
#define HEADS 16
#define HDIM  64
#define SEQ   2048
#define BATCH 2
#define MROWS (BATCH * SEQ)   // 4096

#ifndef __has_builtin
#define __has_builtin(x) 0
#endif

#if defined(__HIP_DEVICE_COMPILE__) && __has_builtin(__builtin_amdgcn_global_load_async_to_lds_b128)
#define HAVE_ASYNC_LDS 1
#else
#define HAVE_ASYNC_LDS 0
#endif

// ---------------------------------------------------------------------------
// WMMA helper: D = A(16x32 bf16) * B(32x16 bf16) + C(16x16 f32), wave32
// ---------------------------------------------------------------------------
__device__ __forceinline__ v8f wmma_bf16(v16bf a, v16bf b, v8f c) {
  return __builtin_amdgcn_wmma_f32_16x16x32_bf16(
      /*neg_a=*/false, a, /*neg_b=*/false, b,
      /*c_mod=*/(short)0, c, /*reuse_a=*/false, /*reuse_b=*/false);
}

// B-operand fragment: lane holds column N = lane%16; elements 0..7 are K=base..base+7,
// elements 8..15 are K=base+16..base+23, base = (lane>=16 ? 8 : 0). Caller passes pointer to that K base.
__device__ __forceinline__ v16bf ld_b16(const bf16_t* p) {
  v8bf lo = *(const v8bf*)(p);
  v8bf hi = *(const v8bf*)(p + 16);
  v16bf r;
#pragma unroll
  for (int i = 0; i < 8; ++i) { r[i] = lo[i]; r[i + 8] = hi[i]; }
  return r;
}

// A-operand fragment: lane holds row M = lane%16; elements 0..7 are K=base..base+7,
// elements 8..15 are K=base+16..base+23, where base = (lane>=16 ? 8 : 0) (+k0).
__device__ __forceinline__ v16bf ld_a16(const bf16_t* p) {
  v8bf lo = *(const v8bf*)(p);
  v8bf hi = *(const v8bf*)(p + 16);
  v16bf r;
#pragma unroll
  for (int i = 0; i < 8; ++i) { r[i] = lo[i]; r[i + 8] = hi[i]; }
  return r;
}

// ---------------------------------------------------------------------------
// 16-byte global -> LDS stage: async (ASYNCcnt) when available, sync fallback
// ---------------------------------------------------------------------------
__device__ __forceinline__ void stage16(const bf16_t* g, bf16_t* l) {
#if HAVE_ASYNC_LDS
  __builtin_amdgcn_global_load_async_to_lds_b128(
      (__attribute__((address_space(1))) v4i*)g,
      (__attribute__((address_space(3))) v4i*)l, 0, 0);
#else
  *(v8bf*)l = *(const v8bf*)g;
#endif
}

__device__ __forceinline__ void wait_async_lds() {
#if HAVE_ASYNC_LDS
#if __has_builtin(__builtin_amdgcn_s_wait_asynccnt)
  __builtin_amdgcn_s_wait_asynccnt(0);
#else
  asm volatile("s_wait_asynccnt 0" ::: "memory");
#endif
#endif
}

// ---------------------------------------------------------------------------
// fp32 -> bf16 conversion / weight transpose
// ---------------------------------------------------------------------------
__global__ void k_cvt_bf16(const float* __restrict__ in, bf16_t* __restrict__ out, int n) {
  int i = blockIdx.x * blockDim.x + threadIdx.x;
  if (i < n) out[i] = (bf16_t)in[i];
}

__global__ void k_transpose_bf16(const float* __restrict__ W, bf16_t* __restrict__ Wt) {
  int i = blockIdx.x * blockDim.x + threadIdx.x;   // over E*E
  int n = i >> 10, k = i & (EMBED - 1);
  if (i < EMBED * EMBED) Wt[i] = (bf16_t)W[k * EMBED + n];   // Wt[n][k] = W[k][n]
}

// ---------------------------------------------------------------------------
// GEMM: block = 128 rows x one 64-wide N panel; 8 waves, each owns a 16-row
// M tile.  B panel chunk (64x32 bf16 = 4KB) staged in LDS via async-to-LDS,
// double-buffered, shared by all 8 waves (8x B reuse).  One barrier per step.
// MODE 0: RoPE epilogue, store bf16 [B,H,S,Dh]      (Q and K projections)
// MODE 2: store bf16 transposed [B,H,Dh,S]          (V projection)
// MODE 3: store fp32 [M,N]                          (output projection)
// ---------------------------------------------------------------------------
template <int MODE>
__global__ __launch_bounds__(256, 1)
void k_gemm(const bf16_t* __restrict__ A, const bf16_t* __restrict__ Bt,
            void* __restrict__ Out) {
  __shared__ __align__(16) bf16_t ldsB[2][64 * 32];   // 2 x 4KB

  const int lane = threadIdx.x & 31;
  const int wv   = threadIdx.x >> 5;
  const int mblk = blockIdx.x >> 4;     // 32 blocks of 128 rows
  const int nt64 = blockIdx.x & 15;     // 64-wide N panel == head index
  const int mt   = mblk * 8 + wv;       // this wave's 16-row M tile
  const int ln   = lane & 15;
  const int hb   = (lane >> 4) << 3;    // 0 or 8 (upper half-wave)
  const int bko  = (lane < 16) ? 0 : 16;

  // cooperative stage mapping: thread t copies 16B of the 64x32 panel chunk
  const int tn = threadIdx.x >> 2;            // 0..63  local n
  const int tk = (threadIdx.x & 3) * 8;       // 0,8,16,24
  const bf16_t* bsrc = Bt + (size_t)(nt64 * 64 + tn) * EMBED + tk;

  stage16(bsrc, &ldsB[0][tn * 32 + tk]);      // prologue: chunk 0

  v8f acc[4];
#pragma unroll
  for (int i = 0; i < 4; ++i) { v8f z = {}; acc[i] = z; }

  const bf16_t* arow = A + (size_t)(mt * 16 + ln) * EMBED;

  for (int k0 = 0; k0 < EMBED; k0 += 32) {
    const int cur = (k0 >> 5) & 1;
    wait_async_lds();
    __syncthreads();                          // chunk `cur` ready for everyone
    if (k0 + 32 < EMBED)                      // prefetch next chunk
      stage16(bsrc + k0 + 32, &ldsB[cur ^ 1][tn * 32 + tk]);

    v16bf a = ld_a16(arow + k0 + hb);
#pragma unroll
    for (int nt = 0; nt < 4; ++nt) {
      v16bf b = ld_b16(&ldsB[cur][(nt * 16 + ln) * 32 + hb]);
      acc[nt] = wmma_bf16(a, b, acc[nt]);
    }
  }

  if (MODE == 3) {                       // fp32 result matrix
    float* o = (float*)Out;
#pragma unroll
    for (int nt = 0; nt < 4; ++nt)
#pragma unroll
      for (int r = 0; r < 8; ++r)
        o[(size_t)(mt * 16 + r + hb) * EMBED + nt64 * 64 + nt * 16 + ln] = acc[nt][r];
  } else if (MODE == 2) {                // V, transposed [B,H,Dh,S]
    bf16_t* vb = (bf16_t*)Out;
    const int h = nt64;
#pragma unroll
    for (int r = 0; r < 8; ++r) {
      const int m = mt * 16 + r + hb;
      const int bq = m >> 11, s = m & (SEQ - 1);
#pragma unroll
      for (int nt = 0; nt < 4; ++nt) {
        const int d = nt * 16 + ln;
        vb[((size_t)(bq * HEADS + h) * HDIM + d) * SEQ + s] = (bf16_t)acc[nt][r];
      }
    }
  } else {                               // Q/K with fused RoPE, [B,H,S,Dh]
    bf16_t* qb = (bf16_t*)Out;
    const int h = nt64;
#pragma unroll
    for (int r = 0; r < 8; ++r) {
      const int m = mt * 16 + r + hb;
      const int bq = m >> 11, s = m & (SEQ - 1);
      const size_t base = ((size_t)(bq * HEADS + h) * SEQ + s) * HDIM;
#pragma unroll
      for (int nt = 0; nt < 2; ++nt) {
        const int d = nt * 16 + ln;                        // 0..31 (pair at d+32)
        const float ang = (float)s * __powf(10000.0f, -(float)d * (1.0f / 32.0f));
        float sn, cs;
        __sincosf(ang, &sn, &cs);
        const float x1 = acc[nt][r], x2 = acc[nt + 2][r];
        qb[base + d]      = (bf16_t)(x1 * cs - x2 * sn);
        qb[base + d + 32] = (bf16_t)(x2 * cs + x1 * sn);
      }
    }
  }
}

// ---------------------------------------------------------------------------
// Flash attention: one wave per 16-row q tile; online softmax; WMMA for
// QK^T and P*V.  P transits LDS (D-layout -> A-layout).  Row-sums of P are
// computed with an extra WMMA against a ones matrix; only the row-max uses
// lane shuffles.  __launch_bounds__(256,1) keeps all fragments in VGPRs
// (no scratch spills of the Q fragments in the 64-iteration flash loop).
// ---------------------------------------------------------------------------
__global__ __launch_bounds__(256, 1)
void k_attn(const bf16_t* __restrict__ qb, const bf16_t* __restrict__ kb,
            const bf16_t* __restrict__ vb, bf16_t* __restrict__ attn) {
  __shared__ __align__(16) bf16_t ldsP[8][16 * 32];   // 1 KB per wave

  const int lane = threadIdx.x & 31;
  const int wv   = threadIdx.x >> 5;
  const int t    = blockIdx.x * 8 + wv;
  const int QT   = SEQ / 16;            // 128 q tiles
  const int qt   = t % QT;
  const int bh   = t / QT;              // 0..B*H-1
  const int ln   = lane & 15;
  const int hb   = (lane >> 4) << 3;    // 0 or 8
  const int bko  = (lane < 16) ? 0 : 16;

  const bf16_t* qh = qb + (size_t)bh * SEQ * HDIM;
  const bf16_t* kh = kb + (size_t)bh * SEQ * HDIM;
  const bf16_t* vh = vb + (size_t)bh * HDIM * SEQ;

  const bf16_t* qrow = qh + (size_t)(qt * 16 + ln) * HDIM;
  const v16bf qa0 = ld_a16(qrow + hb);         // K = 0..31 of Dh
  const v16bf qa1 = ld_a16(qrow + 32 + hb);    // K = 32..63 of Dh

  v16bf ones;                                   // B-operand of all 1.0
#pragma unroll
  for (int i = 0; i < 16; ++i) ones[i] = (bf16_t)1.0f;

  float mrow[8], lrow[8];
  v8f acc[4];
#pragma unroll
  for (int r = 0; r < 8; ++r) { mrow[r] = -3.0e38f; lrow[r] = 0.0f; }
#pragma unroll
  for (int i = 0; i < 4; ++i) { v8f z = {}; acc[i] = z; }

  for (int c = 0; c < SEQ / 32; ++c) {
    const int s0 = c * 32;
    if (c + 1 < SEQ / 32)   // warm L2/WGP$ for next K chunk
      __builtin_prefetch(kh + (size_t)(s0 + 32 + ln) * HDIM, 0, 1);

    // ---- scores: S = (Q K^T) * 1/sqrt(Dh), two 16x16 n-tiles ----
    v8f sc[2];
#pragma unroll
    for (int nt = 0; nt < 2; ++nt) {
      const bf16_t* krow = kh + (size_t)(s0 + nt * 16 + ln) * HDIM;
      v16bf b0 = ld_b16(krow + hb);
      v16bf b1 = ld_b16(krow + 32 + hb);
      v8f z = {};
      z = wmma_bf16(qa0, b0, z);
      z = wmma_bf16(qa1, b1, z);
#pragma unroll
      for (int r = 0; r < 8; ++r) z[r] *= 0.125f;   // 1/sqrt(64)
      sc[nt] = z;
    }

    // ---- online softmax: row max across 16-lane halves ----
    float mc[8];
#pragma unroll
    for (int r = 0; r < 8; ++r) mc[r] = fmaxf(sc[0][r], sc[1][r]);
#pragma unroll
    for (int off = 8; off >= 1; off >>= 1)
#pragma unroll
      for (int r = 0; r < 8; ++r) mc[r] = fmaxf(mc[r], __shfl_xor(mc[r], off, 32));

    float alpha[8];
#pragma unroll
    for (int r = 0; r < 8; ++r) {
      float mn = fmaxf(mrow[r], mc[r]);
      alpha[r] = __expf(mrow[r] - mn);
      mrow[r] = mn;
    }

    // ---- P = exp(S - m) as bf16, staged through LDS ----
#pragma unroll
    for (int nt = 0; nt < 2; ++nt)
#pragma unroll
      for (int r = 0; r < 8; ++r) {
        float p = __expf(sc[nt][r] - mrow[r]);
        // D-layout element (row r+hb, col nt*16+ln) -> LDS row-major 16x32
        ldsP[wv][(r + hb) * 32 + nt * 16 + ln] = (bf16_t)p;
      }

    // rescale running accumulator
#pragma unroll
    for (int nd = 0; nd < 4; ++nd)
#pragma unroll
      for (int r = 0; r < 8; ++r) acc[nd][r] *= alpha[r];

    // make P stores visible before cross-lane A-layout reads (same-wave LDS)
    asm volatile("s_wait_dscnt 0" ::: "memory");
    v16bf pa = ld_a16(&ldsP[wv][ln * 32 + hb]);

    // ---- row sums of P via WMMA against ones (D[r] = sum_k P[row,k]) ----
    v8f rz = {};
    v8f rsv = wmma_bf16(pa, ones, rz);
#pragma unroll
    for (int r = 0; r < 8; ++r) lrow[r] = lrow[r] * alpha[r] + rsv[r];

    // ---- O += P * V, four 16-wide d tiles ----
#pragma unroll
    for (int nd = 0; nd < 4; ++nd) {
      const bf16_t* vrow = vh + (size_t)(nd * 16 + ln) * SEQ + s0 + hb;
      v16bf bv = ld_b16(vrow);
      acc[nd] = wmma_bf16(pa, bv, acc[nd]);
    }
  }

  // ---- normalize and write attn output as bf16 [B,S,E] ----
  const int b = bh >> 4, h = bh & 15;
#pragma unroll
  for (int r = 0; r < 8; ++r) {
    const float inv = 1.0f / lrow[r];
    const size_t mi = (size_t)b * SEQ + qt * 16 + r + hb;
#pragma unroll
    for (int nt = 0; nt < 4; ++nt)
      attn[mi * EMBED + h * HDIM + nt * 16 + ln] = (bf16_t)(acc[nt][r] * inv);
  }
}

// ---------------------------------------------------------------------------
// Host launcher
// ---------------------------------------------------------------------------
extern "C" void kernel_launch(void* const* d_in, const int* in_sizes, int n_in,
                              void* d_out, int out_size, void* d_ws, size_t ws_size,
                              hipStream_t stream) {
  (void)in_sizes; (void)n_in; (void)out_size; (void)ws_size;
  const float* q  = (const float*)d_in[0];
  const float* k  = (const float*)d_in[1];
  const float* v  = (const float*)d_in[2];
  const float* Wq = (const float*)d_in[3];
  const float* Wk = (const float*)d_in[4];
  const float* Wv = (const float*)d_in[5];
  const float* Wo = (const float*)d_in[6];

  char* ws = (char*)d_ws;
  const size_t SZ_X = (size_t)MROWS * EMBED * sizeof(bf16_t);   // 8 MB
  const size_t SZ_W = (size_t)EMBED * EMBED * sizeof(bf16_t);   // 2 MB
  bf16_t* Xq  = (bf16_t*)(ws);
  bf16_t* Xk  = (bf16_t*)(ws + SZ_X);
  bf16_t* Xv  = (bf16_t*)(ws + 2 * SZ_X);
  bf16_t* WqT = (bf16_t*)(ws + 3 * SZ_X);
  bf16_t* WkT = (bf16_t*)(ws + 3 * SZ_X + SZ_W);
  bf16_t* WvT = (bf16_t*)(ws + 3 * SZ_X + 2 * SZ_W);
  bf16_t* WoT = (bf16_t*)(ws + 3 * SZ_X + 3 * SZ_W);
  bf16_t* Qh  = (bf16_t*)(ws + 3 * SZ_X + 4 * SZ_W);
  bf16_t* Kh  = (bf16_t*)(ws + 4 * SZ_X + 4 * SZ_W);
  bf16_t* Vh  = (bf16_t*)(ws + 5 * SZ_X + 4 * SZ_W);
  bf16_t* At  = Xq;   // Xq is dead after the Q projection; reuse for attn out

  const int NX = MROWS * EMBED;              // 4 Mi elements
  k_cvt_bf16<<<NX / 256, 256, 0, stream>>>(q, Xq, NX);
  k_cvt_bf16<<<NX / 256, 256, 0, stream>>>(k, Xk, NX);
  k_cvt_bf16<<<NX / 256, 256, 0, stream>>>(v, Xv, NX);

  const int NW = EMBED * EMBED;              // 1 Mi elements
  k_transpose_bf16<<<NW / 256, 256, 0, stream>>>(Wq, WqT);
  k_transpose_bf16<<<NW / 256, 256, 0, stream>>>(Wk, WkT);
  k_transpose_bf16<<<NW / 256, 256, 0, stream>>>(Wv, WvT);
  k_transpose_bf16<<<NW / 256, 256, 0, stream>>>(Wo, WoT);

  const int GB = (MROWS / 128) * (EMBED / 64);      // 512 blocks, 8 waves each
  k_gemm<0><<<GB, 256, 0, stream>>>(Xq, WqT, (void*)Qh);   // Q proj + RoPE
  k_gemm<0><<<GB, 256, 0, stream>>>(Xk, WkT, (void*)Kh);   // K proj + RoPE
  k_gemm<2><<<GB, 256, 0, stream>>>(Xv, WvT, (void*)Vh);   // V proj, transposed

  const int AB = (BATCH * HEADS * (SEQ / 16)) / 8;  // 512 blocks
  k_attn<<<AB, 256, 0, stream>>>(Qh, Kh, Vh, At);

  k_gemm<3><<<GB, 256, 0, stream>>>(At, WoT, d_out);       // output projection
}
